// _GRUPropagation_48455821033925
// MI455X (gfx1250) — matrix operand order repaired and checked by reading, not compiled, not run on hardware
//
#include <hip/hip_runtime.h>
#include <hip/hip_bf16.h>

// Sizes fixed by the reference
#define NNODE 100000
#define NEDGE 1000000
#define HID   64
#define EDIM  32
#define NSTEP 4

typedef __attribute__((ext_vector_type(16))) _Float16 v16h;
typedef __attribute__((ext_vector_type(8)))  float    v8f;

// ---------------------------------------------------------------------------
// WMMA wrapper: D = A(16x32 f16) x B(32x16 f16) + C(16x16 f32)
// ---------------------------------------------------------------------------
__device__ __forceinline__ v8f wmma_f16(v16h a, v16h b, v8f c) {
  return __builtin_amdgcn_wmma_f32_16x16x32_f16(
      /*neg_a=*/false, a, /*neg_b=*/false, b,
      /*c_mod=*/(short)0, c, /*reuse_a=*/false, /*reuse_b=*/false);
}

__device__ __forceinline__ v8f bcast8(float x) {
  v8f r = {x, x, x, x, x, x, x, x};
  return r;
}

__device__ __forceinline__ float sigmoidf_(float x) {
  return 1.0f / (1.0f + __expf(-x));
}

// ---------------------------------------------------------------------------
// A-fragment (16x32, f16) built from row-major f32 rows [row0..row0+15][ld].
// ISA 16-bit A layout: lane<16 -> M=lane, VGPR0..3 = K 0..7, VGPR4..7 = K 16..23
//                      lane>=16 -> M=lane-16, K 8..15 and K 24..31.
// kbase selects the 32-wide K chunk.
// ---------------------------------------------------------------------------
__device__ __forceinline__ v16h a_frag_f32(const float* __restrict__ base,
                                           int row0, int ld, int kbase, int lane) {
  const int m  = lane & 15;
  const int hs = (lane >> 4) * 8;
  const float* p = base + (size_t)(row0 + m) * ld + kbase;
  v16h a;
#pragma unroll
  for (int j = 0; j < 8; ++j) {
    a[j]     = (_Float16)p[hs + j];
    a[8 + j] = (_Float16)p[16 + hs + j];
  }
  return a;
}

__device__ __forceinline__ v16h a_frag_f16(const _Float16* __restrict__ base,
                                           int row0, int ld, int kbase, int lane) {
  const int m  = lane & 15;
  const int hs = (lane >> 4) * 8;
  const _Float16* p = base + (size_t)(row0 + m) * ld + kbase;
  v16h a;
#pragma unroll
  for (int j = 0; j < 8; ++j) {
    a[j]     = p[hs + j];
    a[8 + j] = p[16 + hs + j];
  }
  return a;
}

// ---------------------------------------------------------------------------
// B-fragment (32x16, f16) for W^T from row-major f16 weights W[out][K] in LDS.
// B[k][n] = W[nrow][k]. B layout: lane<16 -> N=lane, K 0..7 and K 16..23;
// lane>=16 -> N=lane-16, K 8..15 and K 24..31 (within the 32-wide chunk).
// Caller passes nrow = weight-row for this lane's N, kbase = K chunk base.
// ---------------------------------------------------------------------------
__device__ __forceinline__ v16h b_frag_f16(const _Float16* __restrict__ w,
                                           int nrow, int ldk, int kbase, int lane) {
  const _Float16* p = w + (size_t)nrow * ldk + kbase + ((lane >> 4) * 8);
  v16h b;
#pragma unroll
  for (int k = 0; k < 8; ++k) { b[k] = p[k]; b[8 + k] = p[16 + k]; }
  return b;
}

// ===========================================================================
// Kernel 1: msg_base = relu(edge_attr @ W1^T + b1) @ W2^T + b2    [E, 64]
// 128 threads = 4 waves; each wave computes one 16-edge tile with WMMA.
// ===========================================================================
__global__ __launch_bounds__(128) void edge_mlp_kernel(
    const float* __restrict__ edge_attr,
    const float* __restrict__ W1, const float* __restrict__ b1,
    const float* __restrict__ W2, const float* __restrict__ b2,
    float* __restrict__ msg_base) {
  __shared__ _Float16 sW1[HID * EDIM];      // [64][32] f16
  __shared__ _Float16 sW2[HID * HID];       // [64][64] f16
  __shared__ _Float16 sHid[4][16 * HID];    // per-wave 16x64 hidden staging

  const int tid = threadIdx.x;
  for (int i = tid; i < HID * EDIM; i += 128) sW1[i] = (_Float16)W1[i];
  for (int i = tid; i < HID * HID;  i += 128) sW2[i] = (_Float16)W2[i];
  __syncthreads();

  const int lane = tid & 31;
  const int wave = tid >> 5;
  const int e0   = (blockIdx.x * 4 + wave) * 16;  // 16-edge tile base
  const int ln   = lane & 15;                     // N / M index within tile
  const int hi   = lane >> 4;

  // ---- stage 1: hidden = relu(edge_attr @ W1^T + b1), 16x64, K = 32 ----
  v16h a = a_frag_f32(edge_attr, e0, EDIM, 0, lane);
  _Float16* hid = sHid[wave];
#pragma unroll
  for (int t = 0; t < 4; ++t) {                   // 4 column tiles of 16
    v8f c = bcast8(b1[t * 16 + ln]);              // bias depends only on N
    v16h b = b_frag_f16(sW1, t * 16 + ln, EDIM, 0, lane);
    c = wmma_f16(a, b, c);
#pragma unroll
    for (int v = 0; v < 8; ++v) {                 // relu + C-layout -> LDS row-major
      float x = c[v] > 0.0f ? c[v] : 0.0f;
      const int m = v + 8 * hi;
      hid[m * HID + t * 16 + ln] = (_Float16)x;
    }
  }
  __syncthreads();  // lanes read rows written by other lanes of the same wave

  // ---- stage 2: msg_base = hidden @ W2^T + b2, 16x64, K = 64 ----
  v16h a0 = a_frag_f16(hid, 0, HID, 0,  lane);
  v16h a1 = a_frag_f16(hid, 0, HID, 32, lane);
#pragma unroll
  for (int t = 0; t < 4; ++t) {
    v8f c = bcast8(b2[t * 16 + ln]);
    c = wmma_f16(a0, b_frag_f16(sW2, t * 16 + ln, HID, 0,  lane), c);
    c = wmma_f16(a1, b_frag_f16(sW2, t * 16 + ln, HID, 32, lane), c);
#pragma unroll
    for (int v = 0; v < 8; ++v) {
      const int m = v + 8 * hi;
      msg_base[(size_t)(e0 + m) * HID + t * 16 + ln] = c[v];
    }
  }
}

// ===========================================================================
// Kernel 2: zero the aggregation buffer (float4 stores)
// ===========================================================================
__global__ __launch_bounds__(256) void zero_kernel(float4* __restrict__ p, int n4) {
  const int i = blockIdx.x * 256 + threadIdx.x;
  if (i < n4) p[i] = make_float4(0.f, 0.f, 0.f, 0.f);
}

// ===========================================================================
// Kernel 3: msg = msg_base * h[j]; agg[i] += msg  (segment_sum via atomics).
// 16 threads per edge, float4 per thread: coalesced msg_base stream, coalesced
// 256B-per-lane-group gathers of h rows (h is L2-resident: 25.6 MB << 192 MB L2).
// ===========================================================================
__global__ __launch_bounds__(256) void gather_scatter_kernel(
    const float4* __restrict__ msg_base, const float4* __restrict__ h,
    const int* __restrict__ edge_index, float* __restrict__ agg) {
  const int idx = blockIdx.x * 256 + threadIdx.x;  // E*16 threads exactly
  const int e = idx >> 4;
  const int q = idx & 15;
  const int i = edge_index[e];           // destination (segment id)
  const int j = edge_index[NEDGE + e];   // source node
  const float4 m  = msg_base[(size_t)e * 16 + q];
  const float4 hj = h[(size_t)j * 16 + q];
  float* dst = agg + (size_t)i * HID + q * 4;
  atomicAdd(dst + 0, m.x * hj.x);
  atomicAdd(dst + 1, m.y * hj.y);
  atomicAdd(dst + 2, m.z * hj.z);
  atomicAdd(dst + 3, m.w * hj.w);
}

// ===========================================================================
// Kernel 4: GRUCell. 64 threads = 2 waves; each wave handles 16 nodes.
// gi = agg @ W_ih^T + b_ih ; gh = h @ W_hh^T + b_hh ; gates (r,z,n); blend.
// 48 WMMAs per 16-node tile (6 gate tiles x 2 K-chunks x 4 column tiles).
// ===========================================================================
__global__ __launch_bounds__(64) void gru_kernel(
    const float* __restrict__ agg, const float* __restrict__ hcur,
    const float* __restrict__ W_ih, const float* __restrict__ b_ih,
    const float* __restrict__ W_hh, const float* __restrict__ b_hh,
    float* __restrict__ hout) {
  __shared__ _Float16 sWih[3 * HID * HID];  // 24 KB
  __shared__ _Float16 sWhh[3 * HID * HID];  // 24 KB

  const int tid = threadIdx.x;
  for (int i = tid; i < 3 * HID * HID; i += 64) {
    sWih[i] = (_Float16)W_ih[i];
    sWhh[i] = (_Float16)W_hh[i];
  }
  __syncthreads();

  const int lane = tid & 31;
  const int wave = tid >> 5;
  const int n0   = (blockIdx.x * 2 + wave) * 16;
  const int ln   = lane & 15;
  const int hi   = lane >> 4;

  const v16h ax0 = a_frag_f32(agg,  n0, HID, 0,  lane);
  const v16h ax1 = a_frag_f32(agg,  n0, HID, 32, lane);
  const v16h ah0 = a_frag_f32(hcur, n0, HID, 0,  lane);
  const v16h ah1 = a_frag_f32(hcur, n0, HID, 32, lane);

#pragma unroll
  for (int t = 0; t < 4; ++t) {            // 4 x 16 hidden columns
    const int col = t * 16 + ln;
    // r gate
    v8f ir = bcast8(b_ih[col]);
    v8f hr = bcast8(b_hh[col]);
    ir = wmma_f16(ax0, b_frag_f16(sWih, col, HID, 0,  lane), ir);
    ir = wmma_f16(ax1, b_frag_f16(sWih, col, HID, 32, lane), ir);
    hr = wmma_f16(ah0, b_frag_f16(sWhh, col, HID, 0,  lane), hr);
    hr = wmma_f16(ah1, b_frag_f16(sWhh, col, HID, 32, lane), hr);
    // z gate
    v8f iz = bcast8(b_ih[HID + col]);
    v8f hz = bcast8(b_hh[HID + col]);
    iz = wmma_f16(ax0, b_frag_f16(sWih, HID + col, HID, 0,  lane), iz);
    iz = wmma_f16(ax1, b_frag_f16(sWih, HID + col, HID, 32, lane), iz);
    hz = wmma_f16(ah0, b_frag_f16(sWhh, HID + col, HID, 0,  lane), hz);
    hz = wmma_f16(ah1, b_frag_f16(sWhh, HID + col, HID, 32, lane), hz);
    // n gate
    v8f in_ = bcast8(b_ih[2 * HID + col]);
    v8f hn_ = bcast8(b_hh[2 * HID + col]);
    in_ = wmma_f16(ax0, b_frag_f16(sWih, 2 * HID + col, HID, 0,  lane), in_);
    in_ = wmma_f16(ax1, b_frag_f16(sWih, 2 * HID + col, HID, 32, lane), in_);
    hn_ = wmma_f16(ah0, b_frag_f16(sWhh, 2 * HID + col, HID, 0,  lane), hn_);
    hn_ = wmma_f16(ah1, b_frag_f16(sWhh, 2 * HID + col, HID, 32, lane), hn_);

#pragma unroll
    for (int v = 0; v < 8; ++v) {          // C layout: M = v + 8*hi, N = ln
      const float r = sigmoidf_(ir[v] + hr[v]);
      const float z = sigmoidf_(iz[v] + hz[v]);
      const float n = tanhf(in_[v] + r * hn_[v]);
      const int m = v + 8 * hi;
      const float hv = hcur[(size_t)(n0 + m) * HID + col];
      hout[(size_t)(n0 + m) * HID + col] = (1.0f - z) * n + z * hv;
    }
  }
}

// ===========================================================================
// Host-side launch. Workspace layout (floats):
//   msg_base : E*64   (256 MB)
//   agg      : N*64   (25.6 MB)
//   hA, hB   : N*64 each (ping-pong h buffers)
// ===========================================================================
extern "C" void kernel_launch(void* const* d_in, const int* in_sizes, int n_in,
                              void* d_out, int out_size, void* d_ws, size_t ws_size,
                              hipStream_t stream) {
  (void)in_sizes; (void)n_in; (void)out_size; (void)ws_size;

  const float* h_in      = (const float*)d_in[0];
  const float* edge_attr = (const float*)d_in[1];
  const int*   edge_idx  = (const int*)d_in[2];   // [2, E], row 0 = dst, row 1 = src
  const float* W1   = (const float*)d_in[3];
  const float* b1   = (const float*)d_in[4];
  const float* W2   = (const float*)d_in[5];
  const float* b2   = (const float*)d_in[6];
  const float* W_ih = (const float*)d_in[7];
  const float* b_ih = (const float*)d_in[8];
  const float* W_hh = (const float*)d_in[9];
  const float* b_hh = (const float*)d_in[10];
  float* out = (float*)d_out;

  float* msg_base = (float*)d_ws;
  float* agg = msg_base + (size_t)NEDGE * HID;
  float* hA  = agg + (size_t)NNODE * HID;
  float* hB  = hA  + (size_t)NNODE * HID;

  // Step-invariant edge MLP, computed once. E/16 = 62500 wave-tiles, 4 waves/block.
  edge_mlp_kernel<<<NEDGE / 64, 128, 0, stream>>>(edge_attr, W1, b1, W2, b2, msg_base);

  const float* hc = h_in;
  float* targets[NSTEP] = {hA, hB, hA, out};
  for (int s = 0; s < NSTEP; ++s) {
    float* hn = targets[s];
    zero_kernel<<<(NNODE * HID / 4) / 256, 256, 0, stream>>>((float4*)agg, NNODE * HID / 4);
    gather_scatter_kernel<<<(NEDGE * 16) / 256, 256, 0, stream>>>(
        (const float4*)msg_base, (const float4*)hc, edge_idx, agg);
    gru_kernel<<<NNODE / 32, 64, 0, stream>>>(agg, hc, W_ih, b_ih, W_hh, b_hh, hn);
    hc = hn;
  }
}
